// DoRALinear_40114994545140
// MI455X (gfx1250) — matrix operand order repaired and checked by reading, not compiled, not run on hardware
//
#include <hip/hip_runtime.h>
#include <math.h>

typedef __attribute__((ext_vector_type(16))) __bf16 v16bf;
typedef __attribute__((ext_vector_type(8)))  float  v8f;

#define DORA_SCALING 0.5f   // lora_alpha / r = 128/256
#define LP 40               // LDS row pitch in bf16 elems (80 B, 16 B aligned)

struct Q2 { uint4 a, b; };

// ---- WMMA fragment loaders (per CDNA5 ISA 16-bit A 16x32 / B 32x16 lane maps) ----
// A: lane l holds M = l&15, K in {koff..koff+7, koff+16..koff+23}, koff = (l>>4)*8
__device__ __forceinline__ v16bf frag_a(const __bf16* s, int lane) {
  const int m    = lane & 15;
  const int koff = (lane >> 4) << 3;
  const __bf16* p = s + m * LP + koff;
  Q2 q;
  q.a = *reinterpret_cast<const uint4*>(p);        // K = koff .. koff+7
  q.b = *reinterpret_cast<const uint4*>(p + 16);   // K = koff+16 .. koff+23
  return __builtin_bit_cast(v16bf, q);
}
// B: lane l holds N = l&15, K in {koff..koff+7, koff+16..koff+23}, koff = (l>>4)*8
__device__ __forceinline__ v16bf frag_b(const __bf16* s, int lane) {
  const int n    = lane & 15;
  const int koff = (lane >> 4) << 3;
  const __bf16* p = s + n * LP + koff;
  Q2 q;
  q.a = *reinterpret_cast<const uint4*>(p);        // K = koff .. koff+7
  q.b = *reinterpret_cast<const uint4*>(p + 16);   // K = koff+16 .. koff+23
  return __builtin_bit_cast(v16bf, q);
}

__device__ __forceinline__ void split2(float f, __bf16& h, __bf16& l) {
  h = (__bf16)f;
  l = (__bf16)(f - (float)h);
}

__device__ __forceinline__ v8f wmma_bf16(v16bf a, v16bf b, v8f c) {
  return __builtin_amdgcn_wmma_f32_16x16x32_bf16(false, a, false, b, (short)0, c,
                                                 false, false);
}

// ============================================================================
// Kernel 1: combined = baseW + 0.5*(loraB @ loraA), stored as bf16 hi/lo split.
//   loraB: [D][R] fp32 (A operand, M=D, K=R)
//   loraA: [R][D] fp32 (B operand, K=R, N=D) -> staged transposed into LDS [n][k]
//   Block tile 128x64, wave tile 32x32 (K is only 256; ~3% of total WMMA work).
// ============================================================================
__global__ __launch_bounds__(256) void dora_weight_kernel(
    const float* __restrict__ loraB, const float* __restrict__ loraA,
    const float* __restrict__ baseW,
    __bf16* __restrict__ w_hi, __bf16* __restrict__ w_lo, int D, int R) {
  __shared__ __bf16 Ah[128 * LP], Al[128 * LP];
  __shared__ __bf16 Bh[64 * LP],  Bl[64 * LP];

  const int t = threadIdx.x, lane = t & 31, wave = t >> 5;
  const int wm = wave & 3, wn = wave >> 2;
  const int m0 = blockIdx.y * 128;  // o
  const int n0 = blockIdx.x * 64;   // i

  v8f acc[2][2] = {};

  for (int k0 = 0; k0 < R; k0 += 32) {
    // stage A tile: 128x32 fp32 -> hi/lo bf16, float4 loads
#pragma unroll
    for (int i = 0; i < 4; ++i) {
      int e = t + i * 256;
      int row = e >> 3, c4 = (e & 7) << 2;
      float4 v = *reinterpret_cast<const float4*>(loraB + (size_t)(m0 + row) * R + k0 + c4);
      __bf16 h, l;
      split2(v.x, h, l); Ah[row * LP + c4 + 0] = h; Al[row * LP + c4 + 0] = l;
      split2(v.y, h, l); Ah[row * LP + c4 + 1] = h; Al[row * LP + c4 + 1] = l;
      split2(v.z, h, l); Ah[row * LP + c4 + 2] = h; Al[row * LP + c4 + 2] = l;
      split2(v.w, h, l); Ah[row * LP + c4 + 3] = h; Al[row * LP + c4 + 3] = l;
    }
    // stage B tile: loraA is [k][n]; transpose into LDS [n][k] (coalesced reads)
#pragma unroll
    for (int i = 0; i < 8; ++i) {
      int e = t + i * 256;
      int kk = e >> 6, nn = e & 63;
      float v = loraA[(size_t)(k0 + kk) * D + n0 + nn];
      __bf16 h, l; split2(v, h, l);
      Bh[nn * LP + kk] = h; Bl[nn * LP + kk] = l;
    }
    __syncthreads();

    v16bf ah[2], al[2], bh[2], bl[2];
#pragma unroll
    for (int mf = 0; mf < 2; ++mf) {
      ah[mf] = frag_a(Ah + (wm * 32 + mf * 16) * LP, lane);
      al[mf] = frag_a(Al + (wm * 32 + mf * 16) * LP, lane);
    }
#pragma unroll
    for (int nf = 0; nf < 2; ++nf) {
      bh[nf] = frag_b(Bh + (wn * 32 + nf * 16) * LP, lane);
      bl[nf] = frag_b(Bl + (wn * 32 + nf * 16) * LP, lane);
    }
#pragma unroll
    for (int mf = 0; mf < 2; ++mf)
#pragma unroll
      for (int nf = 0; nf < 2; ++nf) {
        acc[mf][nf] = wmma_bf16(ah[mf], bh[nf], acc[mf][nf]);
        acc[mf][nf] = wmma_bf16(ah[mf], bl[nf], acc[mf][nf]);
        acc[mf][nf] = wmma_bf16(al[mf], bh[nf], acc[mf][nf]);
      }
    __syncthreads();
  }

  // epilogue: add base weight, split to hi/lo bf16, store
#pragma unroll
  for (int mf = 0; mf < 2; ++mf)
#pragma unroll
    for (int nf = 0; nf < 2; ++nf) {
      int n = n0 + wn * 32 + nf * 16 + (lane & 15);
#pragma unroll
      for (int v = 0; v < 8; ++v) {
        int m = m0 + wm * 32 + mf * 16 + v + ((lane >> 4) << 3);
        size_t idx = (size_t)m * D + n;
        float c = baseW[idx] + DORA_SCALING * acc[mf][nf][v];
        __bf16 h, l; split2(c, h, l);
        w_hi[idx] = h; w_lo[idx] = l;
      }
    }
}

// ============================================================================
// Kernel 2: deterministic per-row norm + fused scale vector
//   scale[o] = magnitude[o] * turkish[o] / (||combined[o,:]|| + eps)
// ============================================================================
__global__ __launch_bounds__(256) void dora_scale_kernel(
    const __bf16* __restrict__ w_hi, const __bf16* __restrict__ w_lo,
    const float* __restrict__ magnitude, const float* __restrict__ turkish,
    float* __restrict__ scale, int D) {
  __shared__ float red[256];
  const int o = blockIdx.x;
  float s = 0.f;
  for (int i = threadIdx.x; i < D; i += 256) {
    float c = (float)w_hi[(size_t)o * D + i] + (float)w_lo[(size_t)o * D + i];
    s += c * c;
  }
  red[threadIdx.x] = s;
  __syncthreads();
  for (int off = 128; off > 0; off >>= 1) {
    if (threadIdx.x < off) red[threadIdx.x] += red[threadIdx.x + off];
    __syncthreads();
  }
  if (threadIdx.x == 0)
    scale[o] = magnitude[o] * turkish[o] / (sqrtf(red[0]) + 1e-8f);
}

// ============================================================================
// Kernel 3: main GEMM  out[m,n] = scale[n] * sum_k x[m,k]*W[n,k] + bias[n]
//   M=8192, N=4096, K=4096.
//   Block tile 256x128, BK=32, 8 waves as 4x2, wave tile 64x64 (4x4 fragments,
//   48 v_wmma per wave per k-step). Tile traffic: x*32 + W*32 ~ 6.4 GB, served
//   by the 192MB L2 (W hi/lo = 67MB stays resident; x stripe reused across the
//   32 consecutively-launched n-blocks).
// ============================================================================
__global__ __launch_bounds__(256) void dora_main_gemm(
    const float* __restrict__ x,
    const __bf16* __restrict__ w_hi, const __bf16* __restrict__ w_lo,
    const float* __restrict__ scale, const float* __restrict__ bias,
    float* __restrict__ out, int M, int N, int K) {
  __shared__ __bf16 Xh[256 * LP], Xl[256 * LP];
  __shared__ __bf16 Wh[128 * LP], Wl[128 * LP];

  const int t = threadIdx.x, lane = t & 31, wave = t >> 5;
  const int wm = wave & 3, wn = wave >> 2;
  const int m0 = blockIdx.y * 256, n0 = blockIdx.x * 128;

  v8f acc[4][4] = {};

  for (int k0 = 0; k0 < K; k0 += 32) {
    // stage X: 256x32 fp32 -> hi/lo bf16 (8 float4 per thread) + prefetch next k-tile
#pragma unroll
    for (int i = 0; i < 8; ++i) {
      int e = t + i * 256;
      int row = e >> 3, c4 = (e & 7) << 2;
      const float* gp = x + (size_t)(m0 + row) * K + k0 + c4;
      float4 v = *reinterpret_cast<const float4*>(gp);
      __builtin_prefetch(gp + 32, 0, 3);   // next k-tile, same row -> global_prefetch_b8
      __bf16 h, l;
      split2(v.x, h, l); Xh[row * LP + c4 + 0] = h; Xl[row * LP + c4 + 0] = l;
      split2(v.y, h, l); Xh[row * LP + c4 + 1] = h; Xl[row * LP + c4 + 1] = l;
      split2(v.z, h, l); Xh[row * LP + c4 + 2] = h; Xl[row * LP + c4 + 2] = l;
      split2(v.w, h, l); Xh[row * LP + c4 + 3] = h; Xl[row * LP + c4 + 3] = l;
    }
    // stage W: 128x32 bf16 hi+lo, 16 B vector loads (2 per thread per array)
#pragma unroll
    for (int i = 0; i < 2; ++i) {
      int e = t + i * 256;
      int row = e >> 2, c8 = (e & 3) << 3;
      const __bf16* gh = w_hi + (size_t)(n0 + row) * K + k0 + c8;
      const __bf16* gl = w_lo + (size_t)(n0 + row) * K + k0 + c8;
      *reinterpret_cast<uint4*>(&Wh[row * LP + c8]) = *reinterpret_cast<const uint4*>(gh);
      *reinterpret_cast<uint4*>(&Wl[row * LP + c8]) = *reinterpret_cast<const uint4*>(gl);
      __builtin_prefetch(gh + 32, 0, 3);
      __builtin_prefetch(gl + 32, 0, 3);
    }
    __syncthreads();

    v16bf ah[4], al[4], bh[4], bl[4];
#pragma unroll
    for (int mf = 0; mf < 4; ++mf) {
      ah[mf] = frag_a(Xh + (wm * 64 + mf * 16) * LP, lane);
      al[mf] = frag_a(Xl + (wm * 64 + mf * 16) * LP, lane);
    }
#pragma unroll
    for (int nf = 0; nf < 4; ++nf) {
      bh[nf] = frag_b(Wh + (wn * 64 + nf * 16) * LP, lane);
      bl[nf] = frag_b(Wl + (wn * 64 + nf * 16) * LP, lane);
    }
#pragma unroll
    for (int mf = 0; mf < 4; ++mf)
#pragma unroll
      for (int nf = 0; nf < 4; ++nf) {
        acc[mf][nf] = wmma_bf16(ah[mf], bh[nf], acc[mf][nf]);   // hi*hi
        acc[mf][nf] = wmma_bf16(ah[mf], bl[nf], acc[mf][nf]);   // hi*lo
        acc[mf][nf] = wmma_bf16(al[mf], bh[nf], acc[mf][nf]);   // lo*hi
      }
    __syncthreads();
  }

  // epilogue: fused per-row DoRA scale + bias
#pragma unroll
  for (int nf = 0; nf < 4; ++nf) {
    int n = n0 + wn * 64 + nf * 16 + (lane & 15);
    float sc = scale[n], bv = bias[n];
#pragma unroll
    for (int mf = 0; mf < 4; ++mf)
#pragma unroll
      for (int v = 0; v < 8; ++v) {
        int m = m0 + wm * 64 + mf * 16 + v + ((lane >> 4) << 3);
        out[(size_t)m * N + n] = acc[mf][nf][v] * sc + bv;
      }
  }
}

// ============================================================================
extern "C" void kernel_launch(void* const* d_in, const int* in_sizes, int n_in,
                              void* d_out, int out_size, void* d_ws, size_t ws_size,
                              hipStream_t stream) {
  const float* x         = (const float*)d_in[0];
  const float* baseW     = (const float*)d_in[1];
  const float* bias      = (const float*)d_in[2];
  const float* loraA     = (const float*)d_in[3];
  const float* loraB     = (const float*)d_in[4];
  const float* magnitude = (const float*)d_in[5];
  const float* turkish   = (const float*)d_in[6];
  float* out = (float*)d_out;

  const int D = in_sizes[2];        // 4096
  const int R = in_sizes[3] / D;    // 256
  const int M = in_sizes[0] / D;    // B*S = 8192

  char* ws = (char*)d_ws;
  __bf16* w_hi = (__bf16*)ws;
  __bf16* w_lo = (__bf16*)(ws + (size_t)D * D * sizeof(__bf16));
  float*  scl  = (float*)(ws + 2 * (size_t)D * D * sizeof(__bf16));

  dim3 g1(D / 64, D / 128);
  dora_weight_kernel<<<g1, 256, 0, stream>>>(loraB, loraA, baseW, w_hi, w_lo, D, R);

  dora_scale_kernel<<<D, 256, 0, stream>>>(w_hi, w_lo, magnitude, turkish, scl, D);

  dim3 g2(D / 128, M / 256);
  dora_main_gemm<<<g2, 256, 0, stream>>>(x, w_hi, w_lo, scl, bias, out, M, D, D);
}
